// TransformerEncoder_30545807409532
// MI455X (gfx1250) — compile-verified, matrix operand order checked by reading, not run on hardware
//
#include <hip/hip_runtime.h>
#include <cstdint>
#include <cstddef>

// ---------------------------------------------------------------------------
// TransformerConv x2 (PyG-style, heads=1) for MI455X / gfx1250.
// Dense phase: fused Q/K/V/S projection GEMM — one wave owns a 16x16 output
// tile for all four matrices; the A fragment (f32->f16) is built once per
// k-step and reused across 4 v_wmma_f32_16x16x32_f16 issues.
// Edge phase: L2-resident gathers + atomic segment softmax/aggregation.
// ---------------------------------------------------------------------------

typedef __attribute__((ext_vector_type(16))) _Float16 v16h;
typedef __attribute__((ext_vector_type(8)))  _Float16 v8h;
typedef __attribute__((ext_vector_type(8)))  float    v8f;

// ---- weight prep: W [K x C] f32 row-major  ->  Wt [C x K] f16 --------------
__global__ void tc_prep_wt(const float* __restrict__ W, _Float16* __restrict__ Wt,
                           int K, int C) {
  int i = blockIdx.x * blockDim.x + threadIdx.x;
  if (i >= K * C) return;
  int c = i / K, k = i % K;
  Wt[i] = (_Float16)W[(size_t)k * C + c];
}

// Build B fragment from pre-transposed weights (row = out column, 128 halves).
__device__ __forceinline__ v16h tc_load_bfrag(const _Float16* brow, int kk, int g) {
  const v8h b0 = *(const v8h*)(brow + kk + g * 8);
  const v8h b1 = *(const v8h*)(brow + kk + 16 + g * 8);
  return __builtin_shufflevector(b0, b1, 0,1,2,3,4,5,6,7,8,9,10,11,12,13,14,15);
}

// ---- fused GEMM: {Oq,Ok,Ov,Os}[N x C] = A[N x 128] @ W_{q,k,v,s} + b -------
// Wt packs the four transposed weight matrices contiguously: [4][C][128] f16.
template<int C>
__global__ __launch_bounds__(256)
void tc_gemm_qkvs(const float* __restrict__ A,        // [N x 128] f32
                  const _Float16* __restrict__ Wt,    // [4][C][128] f16 (q,k,v,s)
                  const float* __restrict__ bq, const float* __restrict__ bk,
                  const float* __restrict__ bv, const float* __restrict__ bs,
                  float* __restrict__ Oq, float* __restrict__ Ok,
                  float* __restrict__ Ov, float* __restrict__ Os,
                  int N, int nTiles) {
  const int wave = threadIdx.x >> 5;
  const int lane = threadIdx.x & 31;
  const int tile = blockIdx.x * (blockDim.x >> 5) + wave;
  if (tile >= nTiles) return;                 // wave-uniform: EXEC stays all-ones
  constexpr int nTilesN = C / 16;
  constexpr size_t matStride = (size_t)C * 128;
  const int tm  = tile / nTilesN;
  const int tn  = tile % nTilesN;
  const int g   = lane >> 4;                  // lane half (selects K sub-block)
  const int l15 = lane & 15;

  int arowi = tm * 16 + l15; if (arowi >= N) arowi = N - 1;   // clamp (N%16==0 here)
  const float*    arow = A  + (size_t)arowi * 128;
  const _Float16* brow = Wt + (size_t)(tn * 16 + l15) * 128;  // row in matrix 0 (q)

  v8f accq = {}, acck = {}, accv = {}, accs = {};
#pragma unroll
  for (int kk = 0; kk < 128; kk += 32) {
    // A fragment: lane holds row M=l15; elems 0..7 -> K=kk+g*8.., 8..15 -> K=kk+16+g*8..
    const float4 a0 = *(const float4*)(arow + kk + g * 8);
    const float4 a1 = *(const float4*)(arow + kk + g * 8 + 4);
    const float4 a2 = *(const float4*)(arow + kk + 16 + g * 8);
    const float4 a3 = *(const float4*)(arow + kk + 16 + g * 8 + 4);
    v16h a;
    a[0]=(_Float16)a0.x;  a[1]=(_Float16)a0.y;  a[2]=(_Float16)a0.z;  a[3]=(_Float16)a0.w;
    a[4]=(_Float16)a1.x;  a[5]=(_Float16)a1.y;  a[6]=(_Float16)a1.z;  a[7]=(_Float16)a1.w;
    a[8]=(_Float16)a2.x;  a[9]=(_Float16)a2.y;  a[10]=(_Float16)a2.z; a[11]=(_Float16)a2.w;
    a[12]=(_Float16)a3.x; a[13]=(_Float16)a3.y; a[14]=(_Float16)a3.z; a[15]=(_Float16)a3.w;
    // One B fragment per projection; A reused across all four WMMAs.
    v16h b;
    b = tc_load_bfrag(brow + 0 * matStride, kk, g);
    accq = __builtin_amdgcn_wmma_f32_16x16x32_f16(false, a, false, b, (short)0, accq, false, false);
    b = tc_load_bfrag(brow + 1 * matStride, kk, g);
    acck = __builtin_amdgcn_wmma_f32_16x16x32_f16(false, a, false, b, (short)0, acck, false, false);
    b = tc_load_bfrag(brow + 2 * matStride, kk, g);
    accv = __builtin_amdgcn_wmma_f32_16x16x32_f16(false, a, false, b, (short)0, accv, false, false);
    b = tc_load_bfrag(brow + 3 * matStride, kk, g);
    accs = __builtin_amdgcn_wmma_f32_16x16x32_f16(false, a, false, b, (short)0, accs, false, false);
  }
  const int   ocol = tn * 16 + l15;
  const float vbq = bq[ocol], vbk = bk[ocol], vbv = bv[ocol], vbs = bs[ocol];
#pragma unroll
  for (int r = 0; r < 8; ++r) {
    int orow = tm * 16 + r + 8 * g;           // C/D layout: VGPR r -> M = r + 8*(lane/16)
    if (orow < N) {
      size_t o = (size_t)orow * C + ocol;
      Oq[o] = accq[r] + vbq;
      Ok[o] = acck[r] + vbk;
      Ov[o] = accv[r] + vbv;
      Os[o] = accs[r] + vbs;
    }
  }
}

// ---- edge pass 1: alpha[e] = <q[dst], k[src]> / sqrt(C); segment max -------
template<int C>
__global__ void tc_edge_alpha(const float* __restrict__ Q, const float* __restrict__ Kf,
                              const int* __restrict__ src, const int* __restrict__ dst,
                              float* __restrict__ alpha, unsigned int* __restrict__ amaxU,
                              int E) {
  int wid  = (int)((blockIdx.x * (size_t)blockDim.x + threadIdx.x) >> 5);
  int lane = threadIdx.x & 31;
  if (wid >= E) return;
  int s = src[wid], d = dst[wid];
  float part;
  if (C == 128) {
    float4 qv = *(const float4*)(Q  + (size_t)d * C + lane * 4);
    float4 kv = *(const float4*)(Kf + (size_t)s * C + lane * 4);
    part = qv.x * kv.x + qv.y * kv.y + qv.z * kv.z + qv.w * kv.w;
  } else {  // C == 64
    float2 qv = *(const float2*)(Q  + (size_t)d * C + lane * 2);
    float2 kv = *(const float2*)(Kf + (size_t)s * C + lane * 2);
    part = qv.x * kv.x + qv.y * kv.y;
  }
#pragma unroll
  for (int off = 16; off > 0; off >>= 1) part += __shfl_xor(part, off, 32);
  if (lane == 0) {
    float a = part * (C == 128 ? 0.08838834764831845f /*1/sqrt(128)*/ : 0.125f);
    alpha[wid] = a;
    unsigned int u = __float_as_uint(a);
    u = (u & 0x80000000u) ? ~u : (u | 0x80000000u);   // monotonic float->u32 map
    atomicMax(amaxU + d, u);
  }
}

// ---- edge pass 2: ex = exp(alpha - amax[dst]); denom += ex -----------------
__global__ void tc_edge_exp(const int* __restrict__ dst, float* __restrict__ alpha,
                            const unsigned int* __restrict__ amaxU,
                            float* __restrict__ denom, int E) {
  int e = blockIdx.x * blockDim.x + threadIdx.x;
  if (e >= E) return;
  int d = dst[e];
  unsigned int u  = amaxU[d];
  unsigned int fb = (u & 0x80000000u) ? (u & 0x7FFFFFFFu) : ~u;  // inverse map
  float m  = __uint_as_float(fb);
  float ex = expf(alpha[e] - m);
  alpha[e] = ex;                    // overwrite alpha with ex
  atomicAdd(denom + d, ex);
}

// ---- edge pass 3: Out[dst] += (ex/denom[dst]) * v[src] ---------------------
template<int C>
__global__ void tc_edge_aggr(const float* __restrict__ V,
                             const int* __restrict__ src, const int* __restrict__ dst,
                             const float* __restrict__ exv, const float* __restrict__ denom,
                             float* __restrict__ Out, int E) {
  int wid  = (int)((blockIdx.x * (size_t)blockDim.x + threadIdx.x) >> 5);
  int lane = threadIdx.x & 31;
  if (wid >= E) return;
  int s = src[wid], d = dst[wid];
  float a = exv[wid] / (denom[d] + 1e-16f);
  if (C == 128) {
    float4 vv = *(const float4*)(V + (size_t)s * C + lane * 4);
    float* o = Out + (size_t)d * C + lane * 4;
    atomicAdd(o + 0, vv.x * a); atomicAdd(o + 1, vv.y * a);
    atomicAdd(o + 2, vv.z * a); atomicAdd(o + 3, vv.w * a);
  } else {  // C == 64
    float2 vv = *(const float2*)(V + (size_t)s * C + lane * 2);
    float* o = Out + (size_t)d * C + lane * 2;
    atomicAdd(o + 0, vv.x * a); atomicAdd(o + 1, vv.y * a);
  }
}

// ---- elementwise ReLU ------------------------------------------------------
__global__ void tc_relu(float* __restrict__ h, int n) {
  int i = blockIdx.x * blockDim.x + threadIdx.x;
  if (i < n) { float v = h[i]; h[i] = v > 0.0f ? v : 0.0f; }
}

// ---------------------------------------------------------------------------
extern "C" void kernel_launch(void* const* d_in, const int* in_sizes, int n_in,
                              void* d_out, int out_size, void* d_ws, size_t ws_size,
                              hipStream_t stream) {
  const float* x  = (const float*)d_in[0];
  const int*   ei = (const int*)d_in[1];
  const int N = in_sizes[0] / 128;
  const int E = in_sizes[1] / 2;
  const int* src = ei;
  const int* dst = ei + E;

  // Resolve nested-dict leaf order: insertion (Wq,bq,Wk,bk,...) vs sorted keys
  // (Wk,Wq,Ws,Wv,bk,bq,bs,bv). Detect via in_sizes[4]: 128 => bias => insertion.
  const float *Wq1,*bq1,*Wk1,*bk1,*Wv1,*bv1,*Ws1,*bs1;
  const float *Wq2,*bq2,*Wk2,*bk2,*Wv2,*bv2,*Ws2,*bs2;
  auto f = [&](int i) { return (const float*)d_in[i]; };
  if (n_in >= 19 && in_sizes[4] == 128) {   // insertion order
    Wq1=f(3);  bq1=f(4);  Wk1=f(5);  bk1=f(6);  Wv1=f(7);  bv1=f(8);  Ws1=f(9);  bs1=f(10);
    Wq2=f(11); bq2=f(12); Wk2=f(13); bk2=f(14); Wv2=f(15); bv2=f(16); Ws2=f(17); bs2=f(18);
  } else {                                   // sorted-key flatten
    Wk1=f(3);  Wq1=f(4);  Ws1=f(5);  Wv1=f(6);  bk1=f(7);  bq1=f(8);  bs1=f(9);  bv1=f(10);
    Wk2=f(11); Wq2=f(12); Ws2=f(13); Wv2=f(14); bk2=f(15); bq2=f(16); bs2=f(17); bv2=f(18);
  }

  // Workspace carve (~106 MB total)
  char* p = (char*)d_ws;
  auto alloc = [&](size_t bytes) { char* r = p; p += (bytes + 255) & ~(size_t)255; return r; };
  _Float16* WtL1 = (_Float16*)alloc((size_t)4 * 128 * 128 * 2);  // [4][128][128] q,k,v,s
  _Float16* WtL2 = (_Float16*)alloc((size_t)4 * 64  * 128 * 2);  // [4][ 64][128] q,k,v,s
  float* Qb    = (float*)alloc((size_t)N * 128 * 4);
  float* Kb    = (float*)alloc((size_t)N * 128 * 4);
  float* Vb    = (float*)alloc((size_t)N * 128 * 4);
  float* Hb    = (float*)alloc((size_t)N * 128 * 4);   // layer-1 output (S1 + aggr, relu'd)
  float* alpha = (float*)alloc((size_t)E * 4);
  unsigned int* amaxU = (unsigned int*)alloc((size_t)N * 4);
  float* denom = (float*)alloc((size_t)N * 4);
  float* out   = (float*)d_out;                        // [N x 64]

  const int mTiles  = (N + 15) / 16;
  const int tiles1  = mTiles * (128 / 16);
  const int tiles2  = mTiles * (64 / 16);
  const int gemmB1  = (tiles1 + 7) / 8;                // 8 waves / 256-thread block
  const int gemmB2  = (tiles2 + 7) / 8;
  const int edgeWB  = (E + 7) / 8;                     // wave-per-edge blocks
  const int edgeTB  = (E + 255) / 256;

  // ---- weight prep (packed per layer: q,k,v,s) ----
  const int e1 = 128 * 128, e2 = 128 * 64;
  tc_prep_wt<<<(e1 + 255) / 256, 256, 0, stream>>>(Wq1, WtL1 + 0 * (size_t)e1, 128, 128);
  tc_prep_wt<<<(e1 + 255) / 256, 256, 0, stream>>>(Wk1, WtL1 + 1 * (size_t)e1, 128, 128);
  tc_prep_wt<<<(e1 + 255) / 256, 256, 0, stream>>>(Wv1, WtL1 + 2 * (size_t)e1, 128, 128);
  tc_prep_wt<<<(e1 + 255) / 256, 256, 0, stream>>>(Ws1, WtL1 + 3 * (size_t)e1, 128, 128);
  tc_prep_wt<<<(e2 + 255) / 256, 256, 0, stream>>>(Wq2, WtL2 + 0 * (size_t)e2, 128, 64);
  tc_prep_wt<<<(e2 + 255) / 256, 256, 0, stream>>>(Wk2, WtL2 + 1 * (size_t)e2, 128, 64);
  tc_prep_wt<<<(e2 + 255) / 256, 256, 0, stream>>>(Wv2, WtL2 + 2 * (size_t)e2, 128, 64);
  tc_prep_wt<<<(e2 + 255) / 256, 256, 0, stream>>>(Ws2, WtL2 + 3 * (size_t)e2, 128, 64);

  // =================== Layer 1: TransformerConv(128 -> 128) ===================
  hipMemsetAsync(amaxU, 0, (size_t)N * 4, stream);
  hipMemsetAsync(denom, 0, (size_t)N * 4, stream);

  tc_gemm_qkvs<128><<<gemmB1, 256, 0, stream>>>(x, WtL1, bq1, bk1, bv1, bs1,
                                                Qb, Kb, Vb, Hb, N, tiles1);

  tc_edge_alpha<128><<<edgeWB, 256, 0, stream>>>(Qb, Kb, src, dst, alpha, amaxU, E);
  tc_edge_exp<<<edgeTB, 256, 0, stream>>>(dst, alpha, amaxU, denom, E);
  tc_edge_aggr<128><<<edgeWB, 256, 0, stream>>>(Vb, src, dst, alpha, denom, Hb, E);
  tc_relu<<<((size_t)N * 128 + 255) / 256, 256, 0, stream>>>(Hb, N * 128);

  // =================== Layer 2: TransformerConv(128 -> 64) ====================
  hipMemsetAsync(amaxU, 0, (size_t)N * 4, stream);
  hipMemsetAsync(denom, 0, (size_t)N * 4, stream);

  tc_gemm_qkvs<64><<<gemmB2, 256, 0, stream>>>(Hb, WtL2, bq2, bk2, bv2, bs2,
                                               Qb, Kb, Vb, out, N, tiles2);

  tc_edge_alpha<64><<<edgeWB, 256, 0, stream>>>(Qb, Kb, src, dst, alpha, amaxU, E);
  tc_edge_exp<<<edgeTB, 256, 0, stream>>>(dst, alpha, amaxU, denom, E);
  tc_edge_aggr<64><<<edgeWB, 256, 0, stream>>>(Vb, src, dst, alpha, denom, out, E);

  (void)ws_size; (void)out_size; (void)n_in;
}
